// SlotAttention_86002425135206
// MI455X (gfx1250) — matrix operand order repaired and checked by reading, not compiled, not run on hardware
//
#include <hip/hip_runtime.h>
#include <hip/hip_bf16.h>
#include <math.h>

typedef __bf16 bf16;
typedef __bf16 v16bf __attribute__((ext_vector_type(16)));
typedef __bf16 v8bf  __attribute__((ext_vector_type(8)));
typedef float  v8f   __attribute__((ext_vector_type(8)));

#define NB 32
#define NS 4096
#define ND 512
#define NK 8
#define NITER 3
#define EPSV 1e-5f
#define SCALE 0.04419417382415922f   // 1/sqrt(512)

// ---------------- WMMA fragment loaders (bf16 16x16x32; contiguous k = 16h + s in both A and B) ----------------
// A: 16x32 (MxK). lane m = lane&15; bk=(lane>>4)*16; halves 0-7 -> k=bk..bk+7,
//    halves 8-15 -> k=bk+8..bk+15  => two contiguous 16B chunks.
__device__ __forceinline__ v16bf frag_a(const bf16* base, int ld, int k0) {
  const int lane = threadIdx.x & 31;
  const int m = lane & 15;
  const int bk = (lane >> 4) << 4;
  const bf16* p = base + (size_t)m * ld + k0 + bk;
  v8bf lo = *(const v8bf*)p;
  v8bf hi = *(const v8bf*)(p + 8);
  v16bf a;
#pragma unroll
  for (int i = 0; i < 8; ++i) { a[i] = lo[i]; a[i + 8] = hi[i]; }
  return a;
}
// B: 32x16 (KxN), stored as [N][K] row-major. lane n = lane&15; lanes 0-15 hold
// K=0..15, lanes 16-31 hold K=16..31 => one contiguous 32B chunk per lane.
__device__ __forceinline__ v16bf frag_b(const bf16* base, int ld, int k0) {
  const int lane = threadIdx.x & 31;
  const int n = lane & 15;
  const int kb = (lane >> 4) << 4;
  return *(const v16bf*)(base + (size_t)n * ld + k0 + kb);
}

__device__ __forceinline__ float wave_sum(float v) {
#pragma unroll
  for (int m = 16; m >= 1; m >>= 1) v += __shfl_xor(v, m, 32);
  return v;
}
__device__ __forceinline__ float wave_max(float v) {
#pragma unroll
  for (int m = 16; m >= 1; m >>= 1) v = fmaxf(v, __shfl_xor(v, m, 32));
  return v;
}

// ---------------- small utility kernels ----------------
__global__ void cvt_bf16_kernel(const float* __restrict__ src, bf16* __restrict__ dst, int n) {
  int i = blockIdx.x * blockDim.x + threadIdx.x;
  if (i < n) dst[i] = (bf16)src[i];
}

__global__ void init_slots_kernel(const float* __restrict__ emb, float* __restrict__ slotsF) {
  int i = blockIdx.x * blockDim.x + threadIdx.x;            // NB*NK*ND threads
  slotsF[i] = emb[i & (NK * ND - 1)];
}

// ---------------- fused input LayerNorm + K/V projection (bf16 WMMA) ----------------
// Block = 256 threads (8 waves), tile = 16 rows of the flattened [B*S] dimension.
// k stored row-major [b][s][e]; v stored transposed [b][e][s] (for attn@v B-operand).
__global__ __launch_bounds__(256) void ln_kv_kernel(
    const float* __restrict__ inputs, const float* __restrict__ g, const float* __restrict__ bta,
    const bf16* __restrict__ Wk16, const bf16* __restrict__ Wv16,
    bf16* __restrict__ kmat, bf16* __restrict__ vT) {
  __shared__ bf16 ldsA[16][520];                            // +8 halves pad: kill bank conflicts
  const int w = threadIdx.x >> 5, lane = threadIdx.x & 31;
  const int row0 = blockIdx.x * 16;
  for (int rr = 0; rr < 2; ++rr) {
    const int r = w * 2 + rr;
    const float* x = inputs + (size_t)(row0 + r) * ND;
    float s = 0.f, sq = 0.f, xv[16];
#pragma unroll
    for (int i = 0; i < 16; ++i) { float v = x[lane + 32 * i]; xv[i] = v; s += v; sq += v * v; }
    s = wave_sum(s); sq = wave_sum(sq);
    const float mean = s * (1.f / ND);
    const float rs = rsqrtf(sq * (1.f / ND) - mean * mean + EPSV);
#pragma unroll
    for (int i = 0; i < 16; ++i) {
      const int e = lane + 32 * i;
      ldsA[r][e] = (bf16)((xv[i] - mean) * rs * g[e] + bta[e]);
    }
  }
  __syncthreads();
  const bf16* aBase = &ldsA[0][0];
  for (int j = w; j < 64; j += 8) {                         // 2 outputs x 32 N-tiles
    const int which = j >> 5;                               // 0 = K, 1 = V
    const int ntile = j & 31;
    const bf16* bBase = (which ? Wv16 : Wk16) + (size_t)ntile * 16 * ND;
    v8f acc = {};
#pragma unroll
    for (int kk = 0; kk < 16; ++kk) {
      v16bf a = frag_a(aBase, 520, kk * 32);
      v16bf bm = frag_b(bBase, ND, kk * 32);
      acc = __builtin_amdgcn_wmma_f32_16x16x32_bf16(false, a, false, bm, (short)0, acc, false, false);
    }
    const int n = lane & 15, mh = (lane >> 4) * 8;
#pragma unroll
    for (int gi = 0; gi < 8; ++gi) {
      const int grow = row0 + mh + gi;
      if (which == 0) {
        kmat[(size_t)grow * ND + ntile * 16 + n] = (bf16)acc[gi];
      } else {
        const int b = grow >> 12, sidx = grow & (NS - 1);
        vT[((size_t)b * ND + ntile * 16 + n) * NS + sidx] = (bf16)acc[gi];
      }
    }
  }
}

// ---------------- per-iteration: slots LayerNorm + q projection (tiny, f32) ----------------
__global__ __launch_bounds__(256) void slots_q_kernel(
    const float* __restrict__ slotsF, const float* __restrict__ g, const float* __restrict__ bta,
    const float* __restrict__ Wq, bf16* __restrict__ slots16, bf16* __restrict__ q16) {
  __shared__ float sN[NK][ND];
  const int b = blockIdx.x, t = threadIdx.x;
  const int w = t >> 5, lane = t & 31;
  const float* sb = slotsF + (size_t)b * NK * ND;
  {                                                         // wave w handles slot row w
    const float* x = sb + (size_t)w * ND;
    float s = 0.f, sq = 0.f, xv[16];
#pragma unroll
    for (int i = 0; i < 16; ++i) {
      float v = x[lane + 32 * i]; xv[i] = v; s += v; sq += v * v;
      slots16[(size_t)b * NK * ND + (size_t)w * ND + lane + 32 * i] = (bf16)v;
    }
    s = wave_sum(s); sq = wave_sum(sq);
    const float mean = s * (1.f / ND);
    const float rs = rsqrtf(sq * (1.f / ND) - mean * mean + EPSV);
#pragma unroll
    for (int i = 0; i < 16; ++i) { const int e = lane + 32 * i; sN[w][e] = (xv[i] - mean) * rs * g[e] + bta[e]; }
  }
  __syncthreads();
  for (int o = t; o < NK * ND; o += 256) {
    const int k = o >> 9, e = o & (ND - 1);
    const float* wr = Wq + (size_t)e * ND;
    float acc = 0.f;
    for (int d = 0; d < ND; ++d) acc = fmaf(sN[k][d], wr[d], acc);
    q16[((size_t)b * 16 + k) * ND + e] = (bf16)acc;          // rows 8..15 unused padding
  }
}

// ---------------- per-iteration: scores = scale * q @ k^T (bf16 WMMA) ----------------
__global__ __launch_bounds__(128) void scores_kernel(
    const bf16* __restrict__ q16, const bf16* __restrict__ kmat, float* __restrict__ attnF) {
  const int b = blockIdx.x, grp = blockIdx.y;
  const int w = threadIdx.x >> 5, lane = threadIdx.x & 31;
  const int s0 = (grp * 4 + w) * 16;
  const bf16* aBase = q16 + (size_t)b * 16 * ND;
  const bf16* bBase = kmat + ((size_t)b * NS + s0) * ND;     // row n = s, contiguous in e
  v8f acc = {};
#pragma unroll
  for (int kk = 0; kk < 16; ++kk) {
    v16bf a = frag_a(aBase, ND, kk * 32);
    v16bf bm = frag_b(bBase, ND, kk * 32);
    acc = __builtin_amdgcn_wmma_f32_16x16x32_bf16(false, a, false, bm, (short)0, acc, false, false);
  }
  const int n = lane & 15, mh = (lane >> 4) * 8;
#pragma unroll
  for (int gi = 0; gi < 8; ++gi) {
    const int m = mh + gi;
    if (m < NK) attnF[((size_t)b * NK + m) * NS + s0 + n] = acc[gi] * SCALE;
  }
}

// ---------------- per-iteration: softmax over S (f32 out + bf16 copy) ----------------
__global__ __launch_bounds__(256) void softmax_kernel(float* __restrict__ attnF, bf16* __restrict__ attn16) {
  const int b = blockIdx.x >> 3, k = blockIdx.x & 7;
  float* row = attnF + ((size_t)b * NK + k) * NS;
  bf16* row16 = attn16 + ((size_t)b * 16 + k) * NS;
  const int t = threadIdx.x, w = t >> 5, lane = t & 31;
  __shared__ float redA[8], redB[8];
  float xv[16], mx = -3.4e38f;
#pragma unroll
  for (int i = 0; i < 16; ++i) { xv[i] = row[t + 256 * i]; mx = fmaxf(mx, xv[i]); }
  mx = wave_max(mx);
  if (lane == 0) redA[w] = mx;
  __syncthreads();
#pragma unroll
  for (int i = 0; i < 8; ++i) mx = fmaxf(mx, redA[i]);
  float sm = 0.f;
#pragma unroll
  for (int i = 0; i < 16; ++i) { xv[i] = __expf(xv[i] - mx); sm += xv[i]; }
  sm = wave_sum(sm);
  if (lane == 0) redB[w] = sm;
  __syncthreads();
  sm = 0.f;
#pragma unroll
  for (int i = 0; i < 8; ++i) sm += redB[i];
  const float inv = 1.f / sm;
#pragma unroll
  for (int i = 0; i < 16; ++i) {
    const float v = xv[i] * inv;
    row[t + 256 * i] = v;
    row16[t + 256 * i] = (bf16)v;
  }
}

// ---------------- per-iteration: updates = attn @ v (bf16 WMMA, streams vT) ----------------
__global__ __launch_bounds__(128) void updates_kernel(
    const bf16* __restrict__ attn16, const bf16* __restrict__ vT, bf16* __restrict__ upd16) {
  const int b = blockIdx.x, grp = blockIdx.y;
  const int w = threadIdx.x >> 5, lane = threadIdx.x & 31;
  const int e0 = (grp * 4 + w) * 16;
  const bf16* aBase = attn16 + (size_t)b * 16 * NS;
  const bf16* bBase = vT + ((size_t)b * ND + e0) * NS;       // row n = e, contiguous in s
  v8f acc = {};
  for (int ss = 0; ss < NS / 32; ++ss) {
    __builtin_prefetch(bBase + (size_t)(lane & 15) * NS + (ss + 8) * 32, 0, 1);
    v16bf a = frag_a(aBase, NS, ss * 32);
    v16bf bm = frag_b(bBase, NS, ss * 32);
    acc = __builtin_amdgcn_wmma_f32_16x16x32_bf16(false, a, false, bm, (short)0, acc, false, false);
  }
  const int n = lane & 15, mh = (lane >> 4) * 8;
#pragma unroll
  for (int gi = 0; gi < 8; ++gi) {
    const int m = mh + gi;
    if (m < NK) upd16[((size_t)b * NK + m) * ND + e0 + n] = (bf16)acc[gi];
  }
}

// ---------------- per-iteration: GRU gate GEMMs gx/gh (bf16 WMMA + bias) ----------------
__global__ __launch_bounds__(128) void gru_gates_kernel(
    const bf16* __restrict__ upd16, const bf16* __restrict__ slots16,
    const bf16* __restrict__ Wih16, const bf16* __restrict__ Whh16,
    const float* __restrict__ b_ih, const float* __restrict__ b_hh,
    float* __restrict__ gx, float* __restrict__ gh) {
  const int mtile = blockIdx.x, grp = blockIdx.y;
  const int w = threadIdx.x >> 5, lane = threadIdx.x & 31;
  const int e0 = (grp * 4 + w) * 16;                        // 96 N-tiles over 3D=1536
  const bf16* aX = upd16 + (size_t)mtile * 16 * ND;
  const bf16* aH = slots16 + (size_t)mtile * 16 * ND;
  const bf16* bX = Wih16 + (size_t)e0 * ND;
  const bf16* bH = Whh16 + (size_t)e0 * ND;
  v8f accX = {}, accH = {};
#pragma unroll
  for (int kk = 0; kk < 16; ++kk) {
    v16bf ax = frag_a(aX, ND, kk * 32);
    v16bf bx = frag_b(bX, ND, kk * 32);
    accX = __builtin_amdgcn_wmma_f32_16x16x32_bf16(false, ax, false, bx, (short)0, accX, false, false);
    v16bf ah = frag_a(aH, ND, kk * 32);
    v16bf bh = frag_b(bH, ND, kk * 32);
    accH = __builtin_amdgcn_wmma_f32_16x16x32_bf16(false, ah, false, bh, (short)0, accH, false, false);
  }
  const int n = lane & 15, mh = (lane >> 4) * 8;
  const float bxv = b_ih[e0 + n], bhv = b_hh[e0 + n];
#pragma unroll
  for (int gi = 0; gi < 8; ++gi) {
    const int r = mtile * 16 + mh + gi;
    gx[(size_t)r * (3 * ND) + e0 + n] = accX[gi] + bxv;
    gh[(size_t)r * (3 * ND) + e0 + n] = accH[gi] + bhv;
  }
}

// ---------------- per-iteration: GRU elementwise combine ----------------
__global__ void gru_combine_kernel(const float* __restrict__ gx, const float* __restrict__ gh,
                                   float* __restrict__ slotsF) {
  const int i = blockIdx.x * blockDim.x + threadIdx.x;      // NB*NK*ND threads
  const int r = i >> 9, d = i & (ND - 1);
  const size_t base = (size_t)r * 3 * ND;
  const float xr = gx[base + d], xz = gx[base + ND + d], xn = gx[base + 2 * ND + d];
  const float hr = gh[base + d], hz = gh[base + ND + d], hn = gh[base + 2 * ND + d];
  const float rg = 1.f / (1.f + __expf(-(xr + hr)));
  const float zg = 1.f / (1.f + __expf(-(xz + hz)));
  const float ng = tanhf(xn + rg * hn);
  const float h = slotsF[i];
  slotsF[i] = (1.f - zg) * ng + zg * h;
}

extern "C" void kernel_launch(void* const* d_in, const int* in_sizes, int n_in,
                              void* d_out, int out_size, void* d_ws, size_t ws_size,
                              hipStream_t stream) {
  const float* inputs   = (const float*)d_in[0];
  const float* slot_emb = (const float*)d_in[1];
  const float* ln_in_g  = (const float*)d_in[2];
  const float* ln_in_b  = (const float*)d_in[3];
  const float* ln_s_g   = (const float*)d_in[4];
  const float* ln_s_b   = (const float*)d_in[5];
  const float* Wq       = (const float*)d_in[6];
  const float* Wk       = (const float*)d_in[7];
  const float* Wv       = (const float*)d_in[8];
  const float* W_ih     = (const float*)d_in[9];
  const float* W_hh     = (const float*)d_in[10];
  const float* b_ih     = (const float*)d_in[11];
  const float* b_hh     = (const float*)d_in[12];
  float* out = (float*)d_out;

  char* ws = (char*)d_ws; size_t off = 0;
  auto take = [&](size_t bytes) -> void* {
    off = (off + 255) & ~(size_t)255;
    void* p = ws + off; off += bytes; return p;
  };
  bf16*  kmat    = (bf16*) take((size_t)NB * NS * ND * 2);   // 128 MB: k  [b][s][e]
  bf16*  vT      = (bf16*) take((size_t)NB * ND * NS * 2);   // 128 MB: v^T [b][e][s]
  bf16*  Wk16    = (bf16*) take((size_t)ND * ND * 2);
  bf16*  Wv16    = (bf16*) take((size_t)ND * ND * 2);
  bf16*  Wih16   = (bf16*) take((size_t)3 * ND * ND * 2);
  bf16*  Whh16   = (bf16*) take((size_t)3 * ND * ND * 2);
  bf16*  q16     = (bf16*) take((size_t)NB * 16 * ND * 2);   // 16-row stride (M-pad)
  float* attnF   = (float*)take((size_t)NB * NK * NS * 4);   // output attn, f32
  bf16*  attn16  = (bf16*) take((size_t)NB * 16 * NS * 2);   // 16-row stride (M-pad)
  bf16*  upd16   = (bf16*) take((size_t)NB * NK * ND * 2);   // compact [256][512]
  bf16*  slots16 = (bf16*) take((size_t)NB * NK * ND * 2);   // compact [256][512]
  float* slotsF  = (float*)take((size_t)NB * NK * ND * 4);
  float* gx      = (float*)take((size_t)NB * NK * 3 * ND * 4);
  float* gh      = (float*)take((size_t)NB * NK * 3 * ND * 4);
  (void)ws_size; (void)in_sizes; (void)n_in; (void)out_size;

  cvt_bf16_kernel<<<(ND * ND + 255) / 256, 256, 0, stream>>>(Wk, Wk16, ND * ND);
  cvt_bf16_kernel<<<(ND * ND + 255) / 256, 256, 0, stream>>>(Wv, Wv16, ND * ND);
  cvt_bf16_kernel<<<(3 * ND * ND + 255) / 256, 256, 0, stream>>>(W_ih, Wih16, 3 * ND * ND);
  cvt_bf16_kernel<<<(3 * ND * ND + 255) / 256, 256, 0, stream>>>(W_hh, Whh16, 3 * ND * ND);
  init_slots_kernel<<<NB * NK * ND / 256, 256, 0, stream>>>(slot_emb, slotsF);

  ln_kv_kernel<<<NB * NS / 16, 256, 0, stream>>>(inputs, ln_in_g, ln_in_b, Wk16, Wv16, kmat, vT);

  for (int it = 0; it < NITER; ++it) {
    slots_q_kernel<<<NB, 256, 0, stream>>>(slotsF, ln_s_g, ln_s_b, Wq, slots16, q16);
    scores_kernel<<<dim3(NB, NS / 64), 128, 0, stream>>>(q16, kmat, attnF);
    softmax_kernel<<<NB * NK, 256, 0, stream>>>(attnF, attn16);
    updates_kernel<<<dim3(NB, ND / 64), 128, 0, stream>>>(attn16, vT, upd16);
    gru_gates_kernel<<<dim3(16, 24), 128, 0, stream>>>(upd16, slots16, Wih16, Whh16, b_ih, b_hh, gx, gh);
    gru_combine_kernel<<<NB * NK * ND / 256, 256, 0, stream>>>(gx, gh, slotsF);
  }

  hipMemcpyAsync(out, slotsF, (size_t)NB * NK * ND * 4, hipMemcpyDeviceToDevice, stream);
  hipMemcpyAsync(out + NB * NK * ND, attnF, (size_t)NB * NK * NS * 4, hipMemcpyDeviceToDevice, stream);
}
